// DeformableAlignment_6966436954255
// MI455X (gfx1250) — matrix operand order repaired and checked by reading, not compiled, not run on hardware
//
#include <hip/hip_runtime.h>
#include <hip/hip_bf16.h>

typedef __attribute__((ext_vector_type(16))) __bf16 v16bf;
typedef __attribute__((ext_vector_type(8)))  float  v8f;

#define B_    2
#define CX    64
#define H_    192
#define W_    192
#define HW    (H_*W_)
#define DG_   16
#define CG_   8
#define K_    9
#define COUT_ 128
#define KTOT  1152          // DG_*CG_*K_
#define HC    48
#define WC    48
#define HWC   (HC*WC)
#define C0    160
#define C4    432
#define MTILE 32            // pixels per workgroup (2 WMMA M-subtiles)
#define LRELU_ 0.2f
#define MAXRES 5.0f

static __device__ __forceinline__ int imin(int a, int b) { return a < b ? a : b; }
static __device__ __forceinline__ int imax(int a, int b) { return a > b ? a : b; }
// tanh via the TRANS-unit exp: tanh(x) = 1 - 2/(e^{2x}+1)
static __device__ __forceinline__ float fast_tanh(float v) {
    float e = __expf(2.f * v);
    return 1.f - 2.f / (e + 1.f);
}

// ---------------- Stage 1: fused concat + 4x4 avg pool ----------------
__global__ void pool_extra_k(const float* __restrict__ x, const float* __restrict__ flow,
                             const float* __restrict__ add_feat, float* __restrict__ a0) {
    int idx = blockIdx.x * 256 + threadIdx.x;
    int total = B_ * C0 * HWC;
    if (idx >= total) return;
    int xq = idx % WC; int t = idx / WC;
    int y  = t % HC;   t /= HC;
    int ch = t % C0;   int b = t / C0;
    const float* src;
    if (ch < CX)              src = x        + ((size_t)(b*CX      + ch))            * HW;
    else if (ch < CX + 2*DG_) src = flow     + ((size_t)(b*2*DG_   + (ch-CX)))       * HW;
    else                      src = add_feat + ((size_t)(b*CX      + (ch-CX-2*DG_))) * HW;
    float s = 0.f;
    int yb = y*4, xb = xq*4;
    #pragma unroll
    for (int dy = 0; dy < 4; ++dy)
        #pragma unroll
        for (int dx = 0; dx < 4; ++dx)
            s += src[(yb+dy)*W_ + xb+dx];
    a0[idx] = s * (1.f/16.f);
}

// ---------------- Stage 2: direct 3x3 conv (pad=1) at 48x48 ----------------
__global__ void conv3x3_k(const float* __restrict__ in, const float* __restrict__ w,
                          const float* __restrict__ bias, float* __restrict__ out,
                          int Cin, int Cout, int lrelu) {
    int idx = blockIdx.x * 256 + threadIdx.x;
    int total = B_ * Cout * HWC;
    if (idx >= total) return;
    int xq = idx % WC; int t = idx / WC;
    int y  = t % HC;   t /= HC;
    int co = t % Cout; int b = t / Cout;
    float acc = bias[co];
    const float* wb = w  + (size_t)co * Cin * 9;
    const float* ib = in + (size_t)b  * Cin * HWC;
    for (int ci = 0; ci < Cin; ++ci) {
        const float* ip = ib + ci * HWC;
        const float* wp = wb + ci * 9;
        #pragma unroll
        for (int ky = 0; ky < 3; ++ky) {
            int yy = y + ky - 1;
            if (yy < 0 || yy >= HC) continue;
            #pragma unroll
            for (int kx = 0; kx < 3; ++kx) {
                int xx = xq + kx - 1;
                if (xx < 0 || xx >= WC) continue;
                acc = fmaf(ip[yy*WC + xx], wp[ky*3 + kx], acc);
            }
        }
    }
    out[idx] = (lrelu && acc < 0.f) ? LRELU_ * acc : acc;
}

// ------- Stage 3: cast deform weights to bf16, transposed to [K=1152][N=128] -------
__global__ void wprep_k(const float* __restrict__ w, __bf16* __restrict__ Bwt) {
    int idx = blockIdx.x * 256 + threadIdx.x;
    if (idx >= COUT_ * KTOT) return;
    int o = idx / KTOT, k = idx % KTOT;
    Bwt[(size_t)k * COUT_ + o] = (__bf16)w[idx];
}

// ---------------- Stage 4: deformable conv as bf16 WMMA GEMM ----------------
// Workgroup = 256 threads (8 waves). Tile = 32 pixels x 128 outs.
// Phase A: 4608 (pix,g,k) triples sampled cooperatively -> bf16 A panel in LDS (74 KB).
// Phase B: each wave owns a 16-wide N slice; per 32-wide k-step it loads the B
//          fragment ONCE and reuses it for both 16-pixel M-subtiles (2 wmma/iter),
//          halving the L2 weight-stream vs a 16-pixel tile.
__global__ void __launch_bounds__(256)
deform_wmma_k(const float* __restrict__ x, const float* __restrict__ add_feat,
              const float* __restrict__ flow, const float* __restrict__ a4,
              const __bf16* __restrict__ Bwt, const float* __restrict__ bias,
              float* __restrict__ out) {
    __shared__ __bf16 Ash[MTILE][KTOT + 8];   // +8 pad de-conflicts banks (74,240 B)

    int tile  = blockIdx.x;
    int b     = tile / (HW/MTILE);
    int pbase = (tile % (HW/MTILE)) * MTILE;
    int tid   = threadIdx.x;

    const float* a4b = a4 + (size_t)b * C4 * HWC;

    for (int t = tid; t < MTILE * DG_ * K_; t += 256) {   // exactly 18 iters, all lanes
        int pix = t / (DG_*K_);
        int gk  = t % (DG_*K_);
        int g   = gk / K_;
        int k   = gk % K_;
        int p   = pbase + pix;
        int y   = p / W_, xq = p % W_;

        // bilinear upsample of conv head output (48x48 -> 192x192, half-pixel centers)
        float sy = (y  + 0.5f) * 0.25f - 0.5f;
        float sx = (xq + 0.5f) * 0.25f - 0.5f;
        float fy = floorf(sy), fx = floorf(sx);
        float ty = sy - fy,    tx = sx - fx;
        int y0 = (int)fy, x0 = (int)fx;
        int ya = imax(y0, 0),   yb2 = imin(y0+1, HC-1);
        int xa = imax(x0, 0),   xb2 = imin(x0+1, WC-1);
        float u00 = (1.f-ty)*(1.f-tx), u01 = (1.f-ty)*tx, u10 = ty*(1.f-tx), u11 = ty*tx;
        int i00 = ya*WC+xa, i01 = ya*WC+xb2, i10 = yb2*WC+xa, i11 = yb2*WC+xb2;

        int cdy = g*18 + 2*k, cdx = cdy + 1, cm = 288 + g*K_ + k;
        const float* pdy = a4b + (size_t)cdy * HWC;
        const float* pdx = a4b + (size_t)cdx * HWC;
        const float* pms = a4b + (size_t)cm  * HWC;
        float ody = u00*pdy[i00] + u01*pdy[i01] + u10*pdy[i10] + u11*pdy[i11];
        float odx = u00*pdx[i00] + u01*pdx[i01] + u10*pdx[i10] + u11*pdx[i11];
        float om  = u00*pms[i00] + u01*pms[i01] + u10*pms[i10] + u11*pms[i11];

        // offset = 5*tanh(.) + flow (pair order reversed per reference), mask = sigmoid
        float dy = MAXRES * fast_tanh(ody) + flow[((size_t)(b*2*DG_ + 2*g + 1))*HW + p];
        float dx = MAXRES * fast_tanh(odx) + flow[((size_t)(b*2*DG_ + 2*g    ))*HW + p];
        float msk = 1.f / (1.f + __expf(-om));

        float py = dy + (float)y  + (float)(k/3 - 1);
        float px = dx + (float)xq + (float)(k%3 - 1);
        float fpy = floorf(py), fpx = floorf(px);
        int iy0 = (int)fpy, ix0 = (int)fpx;
        float wy1 = py - fpy, wx1 = px - fpx;
        float c00 = (1.f-wy1)*(1.f-wx1), c01 = (1.f-wy1)*wx1;
        float c10 = wy1*(1.f-wx1),       c11 = wy1*wx1;
        bool vy0 = (iy0   >= 0) && (iy0   < H_);
        bool vy1 = (iy0+1 >= 0) && (iy0+1 < H_);
        bool vx0 = (ix0   >= 0) && (ix0   < W_);
        bool vx1 = (ix0+1 >= 0) && (ix0+1 < W_);
        if (!(vy0 && vx0)) c00 = 0.f;
        if (!(vy0 && vx1)) c01 = 0.f;
        if (!(vy1 && vx0)) c10 = 0.f;
        if (!(vy1 && vx1)) c11 = 0.f;
        int yc0 = imin(imax(iy0,   0), H_-1), yc1 = imin(imax(iy0+1, 0), H_-1);
        int xc0 = imin(imax(ix0,   0), W_-1), xc1 = imin(imax(ix0+1, 0), W_-1);
        int j00 = yc0*W_+xc0, j01 = yc0*W_+xc1, j10 = yc1*W_+xc0, j11 = yc1*W_+xc1;

        // x_input channel (g*8+c): c<4 from x[g*4+c], else add_feat[g*4+c-4]
        #pragma unroll
        for (int c = 0; c < CG_; ++c) {
            const float* src = (c < 4)
                ? x        + ((size_t)(b*CX + g*4 + c    )) * HW
                : add_feat + ((size_t)(b*CX + g*4 + (c-4))) * HW;
            float v = c00*src[j00] + c01*src[j01] + c10*src[j10] + c11*src[j11];
            Ash[pix][(g*CG_ + c)*K_ + k] = (__bf16)(v * msk);
        }
    }
    __syncthreads();

    // ---- WMMA phase: EXEC all ones, uniform control flow ----
    int wave = tid >> 5;
    int lane = tid & 31;
    int ncol = lane & 15;
    int half = lane >> 4;
    int apix0 = lane & 15;        // M-subtile 0: pixels pbase + 0..15
    int apix1 = 16 + (lane & 15); // M-subtile 1: pixels pbase + 16..31

    v8f acc0 = {0.f,0.f,0.f,0.f,0.f,0.f,0.f,0.f};
    v8f acc1 = {0.f,0.f,0.f,0.f,0.f,0.f,0.f,0.f};
    for (int kb = 0; kb < KTOT; kb += 32) {
        v16bf av0, av1, bv;
        // B: lane = N column (ncol), VGPRs sweep K in the same packing as A
        const __bf16* brow = Bwt + (size_t)kb * COUT_ + wave * 16 + ncol;
        #pragma unroll
        for (int v = 0; v < 8; ++v) {
            int kk = kb + ((v < 4) ? 0 : 16) + half*8 + (v & 3)*2;  // 16-bit A K-packing
            av0[2*v]   = Ash[apix0][kk];
            av0[2*v+1] = Ash[apix0][kk+1];
            av1[2*v]   = Ash[apix1][kk];
            av1[2*v+1] = Ash[apix1][kk+1];
            bv[2*v]    = brow[(size_t)(kk - kb) * COUT_];
            bv[2*v+1]  = brow[(size_t)(kk - kb + 1) * COUT_];
        }
        acc0 = __builtin_amdgcn_wmma_f32_16x16x32_bf16(
            false, av0, false, bv, (short)0, acc0, false, false);
        acc1 = __builtin_amdgcn_wmma_f32_16x16x32_bf16(
            false, av1, false, bv, (short)0, acc1, false, false);
    }

    // C/D layout: VGPR r -> M = r + 8*half, N = ncol
    int nglob = wave*16 + ncol;
    float bia = bias[nglob];
    float* ob0 = out + ((size_t)(b*COUT_ + nglob)) * HW + pbase;        // subtile 0
    float* ob1 = ob0 + 16;                                              // subtile 1
    #pragma unroll
    for (int r = 0; r < 8; ++r) {
        ob0[half*8 + r] = acc0[r] + bia;
        ob1[half*8 + r] = acc1[r] + bia;
    }
}

extern "C" void kernel_launch(void* const* d_in, const int* in_sizes, int n_in,
                              void* d_out, int out_size, void* d_ws, size_t ws_size,
                              hipStream_t stream) {
    const float* x    = (const float*)d_in[0];
    const float* flow = (const float*)d_in[1];
    const float* add  = (const float*)d_in[2];
    const float* w1   = (const float*)d_in[3];  const float* b1 = (const float*)d_in[4];
    const float* w2   = (const float*)d_in[5];  const float* b2 = (const float*)d_in[6];
    const float* w3   = (const float*)d_in[7];  const float* b3 = (const float*)d_in[8];
    const float* w4   = (const float*)d_in[9];  const float* b4 = (const float*)d_in[10];
    const float* wt   = (const float*)d_in[11]; const float* bs = (const float*)d_in[12];
    float* out = (float*)d_out;
    float* ws  = (float*)d_ws;

    size_t o = 0;
    float* a0 = ws + o; o += (size_t)B_ * C0 * HWC;   // pooled concat  (2,160,48,48)
    float* a1 = ws + o; o += (size_t)B_ * 64 * HWC;
    float* a2 = ws + o; o += (size_t)B_ * 64 * HWC;
    float* a3 = ws + o; o += (size_t)B_ * 64 * HWC;
    float* a4 = ws + o; o += (size_t)B_ * C4 * HWC;   // head output    (2,432,48,48)
    __bf16* Bwt = (__bf16*)(ws + o);                  // 1152x128 bf16 weights (294 KB)

    dim3 blk(256);
    pool_extra_k<<<(B_*C0*HWC + 255)/256, blk, 0, stream>>>(x, flow, add, a0);
    conv3x3_k  <<<(B_*64*HWC + 255)/256, blk, 0, stream>>>(a0, w1, b1, a1, C0, 64, 1);
    conv3x3_k  <<<(B_*64*HWC + 255)/256, blk, 0, stream>>>(a1, w2, b2, a2, 64, 64, 1);
    conv3x3_k  <<<(B_*64*HWC + 255)/256, blk, 0, stream>>>(a2, w3, b3, a3, 64, 64, 1);
    conv3x3_k  <<<(B_*C4*HWC + 255)/256, blk, 0, stream>>>(a3, w4, b4, a4, 64, C4, 0);
    wprep_k    <<<(COUT_*KTOT + 255)/256, blk, 0, stream>>>(wt, Bwt);
    deform_wmma_k<<<B_*HW/MTILE, blk, 0, stream>>>(x, add, flow, a4, Bwt, bs, out);
}
